// FixedGaussianWindow1d_74388833567038
// MI455X (gfx1250) — matrix operand order repaired and checked by reading, not compiled, not run on hardware
//
#include <hip/hip_runtime.h>

// Problem constants (from reference)
#define TMAX   490      // MIN_TP (time samples)
#define NWIN   460      // windows
#define NBATCH 8
#define CH     100      // channels
#define CPAD   112      // 7 WMMA tiles of 16
#define KPAD   512      // K padded to multiple of 32
#define LDY    520      // LDS row stride in f16 elems (512 + 8 pad -> conflict-free b128 loads)

typedef _Float16 v2h  __attribute__((ext_vector_type(2)));
typedef _Float16 v8h  __attribute__((ext_vector_type(8)));
typedef _Float16 v16h __attribute__((ext_vector_type(16)));
typedef float    v8f  __attribute__((ext_vector_type(8)));

__global__ __launch_bounds__(256, 2)
void win_corr_wmma_kernel(const float* __restrict__ x,     // [B][T][C]
                          const float* __restrict__ Ash,   // [W][T]
                          float* __restrict__ out)         // [B][W][C][C]
{
    extern __shared__ char smem[];
    _Float16* Yt   = (_Float16*)smem;                         // [CPAD][LDY], f16, Y^T (channel-major)
    float*    psum = (float*)(smem + (size_t)CPAD * LDY * 2); // [2][112] partial col sums
    float*    pss  = psum + 2 * CPAD;                         // [2][112] partial col sum-of-squares
    float*    mean = pss  + 2 * CPAD;                         // [112]
    float*    dinv = mean + CPAD;                             // [112] 1/sqrt(var)

    const int tid = threadIdx.x;
    const int w   = blockIdx.x;
    const int b   = blockIdx.y;

    const float* xb = x   + (size_t)b * TMAX * CH;
    const float* aw = Ash + (size_t)w * TMAX;
    float*       ob = out + ((size_t)b * NWIN + w) * (CH * CH);

    // ---- 1) zero Yt (covers K pad cols 490..511 and channel pad rows 100..111)
    {
        v8h z = {};
        v8h* y8 = (v8h*)Yt;
        for (int i = tid; i < (CPAD * LDY) / 8; i += 256) y8[i] = z;
    }
    __syncthreads();

    // ---- 2) fill Yt[c][t] = x[b][t][c] * A[w][t]  (coalesced global reads over c)
    for (int i = tid; i < TMAX * CH; i += 256) {
        int t = i / CH;
        int c = i - t * CH;
        float y = xb[i] * aw[t];
        Yt[c * LDY + t] = (_Float16)y;
    }
    __syncthreads();

    // ---- 3) per-channel partial sums (2 segments x 112 channels), paired f16 loads
    if (tid < 2 * CPAD) {
        int seg = tid / CPAD;                 // 0 or 1
        int c   = tid - seg * CPAD;
        int i0  = seg ? 124 : 0;              // pair index range: [0,124) and [124,245)
        int i1  = seg ? 245 : 124;            // covers t in [0,248) and [248,490)
        const v2h* row2 = (const v2h*)(Yt + c * LDY);
        float s = 0.0f, q = 0.0f;
        for (int i = i0; i < i1; ++i) {
            v2h p = row2[i];
            float v0 = (float)p.x, v1 = (float)p.y;
            s += v0 + v1;
            q += v0 * v0 + v1 * v1;
        }
        psum[seg * CPAD + c] = s;
        pss [seg * CPAD + c] = q;
    }
    __syncthreads();

    // ---- 4) mean + inverse std:  var = sum(y^2) - T*mean^2 = q - s*mean
    if (tid < CPAD) {
        float s = psum[tid] + psum[CPAD + tid];
        float q = pss [tid] + pss [CPAD + tid];
        float m = s * (1.0f / (float)TMAX);
        float var = q - s * m;
        mean[tid] = m;
        dinv[tid] = (tid < CH) ? rsqrtf(fmaxf(var, 1e-20f)) : 0.0f;
    }
    __syncthreads();

    // ---- 5) Gram via WMMA, upper-triangle tiles only (cov is symmetric):
    //         cov = Y^T Y - T m m^T ; corr = clip(cov * dinv_i * dinv_j, -1, 1)
    const int lane = tid & 31;
    const int wave = tid >> 5;          // 8 waves
    const int l15  = lane & 15;
    const int hi   = lane >> 4;         // 0: lanes 0-15, 1: lanes 16-31

    for (int idx = wave; idx < 28; idx += 8) {   // 28 = 7*8/2 upper-triangle tiles
        // map idx -> (tm, tn) with tm <= tn over the 7x7 tile grid
        int tm = 0, off = idx;
        while (off >= 7 - tm) { off -= 7 - tm; ++tm; }
        const int tn = tm + off;

        // A-operand rows = M channels, B-operand rows = N channels (same array: G = Y^T Y)
        const _Float16* Ar = Yt + (tm * 16 + l15) * LDY;
        const _Float16* Br = Yt + (tn * 16 + l15) * LDY;

        v8f acc = {};
        #pragma unroll 4
        for (int k0 = 0; k0 < KPAD; k0 += 32) {
            // 16-bit A layout: lo lanes hold K k0+[0..7] & k0+[16..23]; hi lanes +8
            v8h a0 = *(const v8h*)(Ar + k0 + hi * 8);
            v8h a1 = *(const v8h*)(Ar + k0 + hi * 8 + 16);
            // 16-bit B layout: lo lanes hold K k0+[0..7] & k0+[16..23]; hi lanes +8
            v8h b0 = *(const v8h*)(Br + k0 + hi * 8);
            v8h b1 = *(const v8h*)(Br + k0 + hi * 8 + 16);
            v16h A = __builtin_shufflevector(a0, a1, 0,1,2,3,4,5,6,7,8,9,10,11,12,13,14,15);
            v16h B = __builtin_shufflevector(b0, b1, 0,1,2,3,4,5,6,7,8,9,10,11,12,13,14,15);
            acc = __builtin_amdgcn_wmma_f32_16x16x32_f16(
                /*neg_a=*/false, A, /*neg_b=*/false, B,
                /*c_mod=*/(short)0, acc, /*reuse_a=*/false, /*reuse_b=*/false);
        }

        const int   cN  = tn * 16 + l15;     // C/D layout: N = lane&15
        const float mN  = mean[cN];
        const float dN  = dinv[cN];
        const bool  mirror = (tm != tn);

        if (tn < 6) {
            // fast path: all cM, cN < 96 < CH — no guards, no exec-mask churn
            #pragma unroll
            for (int j = 0; j < 8; ++j) {
                const int cM = tm * 16 + j + hi * 8;   // C/D layout: M = vgpr + 8*hi
                float v = (acc[j] - (float)TMAX * mean[cM] * mN) * (dinv[cM] * dN);
                v = fminf(fmaxf(v, -1.0f), 1.0f);
                ob[cM * CH + cN] = v;
                if (mirror) ob[cN * CH + cM] = v;      // symmetric counterpart
            }
        } else {
            // edge tiles touching the 96..111 pad band
            const bool okN = (cN < CH);
            #pragma unroll
            for (int j = 0; j < 8; ++j) {
                const int cM = tm * 16 + j + hi * 8;
                float v = (acc[j] - (float)TMAX * mean[cM] * mN) * (dinv[cM] * dN);
                v = fminf(fmaxf(v, -1.0f), 1.0f);
                const bool ok = okN && (cM < CH);
                if (ok)           ob[cM * CH + cN] = v;
                if (ok && mirror) ob[cN * CH + cM] = v;
            }
        }
    }
}

extern "C" void kernel_launch(void* const* d_in, const int* in_sizes, int n_in,
                              void* d_out, int out_size, void* d_ws, size_t ws_size,
                              hipStream_t stream) {
    (void)in_sizes; (void)n_in; (void)d_ws; (void)ws_size; (void)out_size;
    const float* x   = (const float*)d_in[0];   // [8][490][100] f32
    const float* Ash = (const float*)d_in[1];   // [460][490]    f32
    float*       out = (float*)d_out;           // [8][460][100][100] f32

    const size_t shmem = (size_t)CPAD * LDY * sizeof(_Float16)   // Yt
                       + (size_t)(2 * CPAD) * sizeof(float) * 2  // psum + pss
                       + (size_t)CPAD * sizeof(float) * 2;       // mean + dinv

    // > 64 KB dynamic LDS: raise the per-kernel cap (WGP has 320 KB).
    (void)hipFuncSetAttribute((const void*)win_corr_wmma_kernel,
                              hipFuncAttributeMaxDynamicSharedMemorySize, (int)shmem);

    dim3 grid(NWIN, NBATCH);
    win_corr_wmma_kernel<<<grid, dim3(256), shmem, stream>>>(x, Ash, out);
}
